// CNNPredictor_4913442586988
// MI455X (gfx1250) — matrix operand order repaired and checked by reading, not compiled, not run on hardware
//
#include <hip/hip_runtime.h>

// ---------------------------------------------------------------------------
// MI455X (gfx1250).
// Factorization: feat@Wh^T = qW[c,:] + ctxW[b,t,:] + [|q-ctx|, q*ctx]@Wh_tail^T
// -> stage-1 WMMA work halves (137 -> 69 GFLOP); qW/ctxW/gW are tiny WMMA
// GEMMs. All weights/activations L2-resident (192MB L2) => matrix-core bound
// => v_wmma_f32_16x16x32_bf16 everywhere. B panels double-buffered in LDS via
// global_load_async_to_lds_b128 + s_wait_asynccnt (CDNA5 async copy path).
// ---------------------------------------------------------------------------

typedef __attribute__((ext_vector_type(16))) __bf16 v16bf;
typedef __attribute__((ext_vector_type(8)))  __bf16 v8bf;
typedef __attribute__((ext_vector_type(8)))  float  v8f;

#define Bsz  8
#define Csz  64
#define Tsz  128
#define Esz  256
#define K4E  1024     // 4*E
#define KH   512      // reduced K (segments |q-ctx|, q*ctx)
#define LDF  520      // feat row stride (bf16): 16B-aligned rows, bank-optimal
#define LDB  520      // B-panel row stride
#define LDH  1032     // h2 row stride (K=1024 + pad)
#define NFsz 128

// round-to-nearest-even f32 -> bf16 (explicit, no libcall dependence)
__device__ __forceinline__ __bf16 f2bf(float f) {
  unsigned u = __builtin_bit_cast(unsigned, f);
  u = (u + 0x7FFFu + ((u >> 16) & 1u)) >> 16;
  unsigned short h = (unsigned short)u;
  return __builtin_bit_cast(__bf16, h);
}

// A-fragment: 16x32 bf16 (MxK). ISA layout:
// lane<16 -> M=lane, K {k0..k0+7, k0+16..k0+23}; lane>=16 -> M=lane-16, +8.
__device__ __forceinline__ v16bf load_frag_a(const __bf16* mat, int ld, int row0, int k0) {
  const int lane = threadIdx.x & 31;
  const int m  = lane & 15;
  const int kh = (lane >> 4) << 3;              // 0 or 8
  const __bf16* p = mat + (size_t)(row0 + m) * ld + k0 + kh;
  v8bf lo = *(const v8bf*)(p);
  v8bf hi = *(const v8bf*)(p + 16);
  v16bf r;
#pragma unroll
  for (int i = 0; i < 8; ++i) { r[i] = lo[i]; r[i + 8] = hi[i]; }
  return r;
}

// B-fragment: 32x16 bf16 (KxN) from W[n][k] row-major: lane holds col n=lane%16,
// K {k0..k0+7, k0+16..k0+23} (lanes 0-15) / +8 (lanes 16-31).
__device__ __forceinline__ v16bf load_frag_b(const __bf16* mat, int ld, int n0, int k0) {
  const int lane = threadIdx.x & 31;
  const int n  = lane & 15;
  const int kh = (lane >> 4) << 3;              // 0 or 8
  const __bf16* p = mat + (size_t)(n0 + n) * ld + k0 + kh;
  v8bf lo = *(const v8bf*)(p);
  v8bf hi = *(const v8bf*)(p + 16);
  v16bf r;
#pragma unroll
  for (int i = 0; i < 8; ++i) { r[i] = lo[i]; r[i + 8] = hi[i]; }
  return r;
}

__device__ __forceinline__ v8f wmma_bf16(v16bf a, v16bf b, v8f c) {
  return __builtin_amdgcn_wmma_f32_16x16x32_bf16(false, a, false, b, (short)0, c,
                                                 false, false);
}

// CDNA5 async global -> LDS copy (ASYNCcnt-tracked), 16B per lane.
__device__ __forceinline__ void async_copy_b128(const void* gptr, void* lptr) {
  unsigned lds = (unsigned)(uintptr_t)lptr;   // flat addr low 32 bits == LDS offset
  asm volatile("global_load_async_to_lds_b128 %0, %1, off"
               :: "v"(lds), "v"(gptr) : "memory");
}
__device__ __forceinline__ void wait_async0() {
#if __has_builtin(__builtin_amdgcn_s_wait_asynccnt)
  __builtin_amdgcn_s_wait_asynccnt(0);
#else
  asm volatile("s_wait_asynccnt 0" ::: "memory");
#endif
}

// ---------------------------------------------------------------------------
// k0: bf16 conversions: W_hidden, W_lin, context, query
// ---------------------------------------------------------------------------
__global__ void __launch_bounds__(256)
k0_convert(const float* __restrict__ Wh, const float* __restrict__ Wl,
           const float* __restrict__ ctx, const float* __restrict__ q,
           __bf16* __restrict__ Whb, __bf16* __restrict__ Wlb,
           __bf16* __restrict__ ctxb, __bf16* __restrict__ qb) {
  int i = blockIdx.x * 256 + threadIdx.x;
  if (i < K4E * K4E)       Whb[i]  = f2bf(Wh[i]);
  if (i < Esz * K4E)       Wlb[i]  = f2bf(Wl[i]);
  if (i < Bsz * Tsz * Esz) ctxb[i] = f2bf(ctx[i]);
  if (i < Csz * Esz)       qb[i]   = f2bf(q[i]);
}

// ---------------------------------------------------------------------------
// kgemm: Out[m][n] = sum_k A[m][k]*B[n][k]; A,B bf16 global, Out f32.
// grid = (M/16, N/128); 8 waves, one 16x16 N-tile per wave.
// ---------------------------------------------------------------------------
__global__ void __launch_bounds__(256)
kgemm(const __bf16* __restrict__ A, int lda,
      const __bf16* __restrict__ Bm, int ldb,
      float* __restrict__ Out, int ldo, int K) {
  const int tid = threadIdx.x, wave = tid >> 5, lane = tid & 31;
  const int m0 = blockIdx.x * 16;
  const int n0 = blockIdx.y * 128 + wave * 16;
  v8f acc = {};
  for (int kt = 0; kt < K / 32; ++kt) {
    v16bf a  = load_frag_a(A, lda, m0, kt * 32);
    v16bf bb = load_frag_b(Bm, ldb, n0, kt * 32);
    acc = wmma_bf16(a, bb, acc);
  }
  int n = n0 + (lane & 15);
  int rbase = (lane >> 4) << 3;
#pragma unroll
  for (int r = 0; r < 8; ++r)
    Out[(size_t)(m0 + rbase + r) * ldo + n] = acc[r];
}

// ---------------------------------------------------------------------------
// k1: per (b,c): score[t] = Wv . tanh(qW[c]+ctxW[b,t]+bh + featK512@WhTail^T),
//     masked softmax, g = attn@ctx. One WG (8 waves) per (b,c).
// LDS: feat[128][LDF] bf16 (133KB) + 2 x Bp[32][LDB] bf16 (66KB, async dbuf).
// ---------------------------------------------------------------------------
__global__ void __launch_bounds__(256)
k1_attn(const float* __restrict__ query, const float* __restrict__ context,
        const int* __restrict__ mask, const __bf16* __restrict__ Wh,
        const float* __restrict__ bh, const float* __restrict__ Wv,
        const float* __restrict__ qW, const float* __restrict__ ctxW,
        float* __restrict__ g_out, __bf16* __restrict__ gb_out) {
  extern __shared__ unsigned char smem[];
  __bf16* feat = (__bf16*)smem;                                   // 128 x LDF
  __bf16* Bp0  = (__bf16*)(smem + (size_t)Tsz * LDF * 2);         // 32 x LDB
  __bf16* Bp1  = Bp0 + (size_t)32 * LDB;                          // 32 x LDB
  float*  sc   = (float*)(Bp1 + (size_t)32 * LDB);                // 128
  float*  red  = sc + Tsz;                                        // 32

  const int tid  = threadIdx.x;
  const int wave = tid >> 5;
  const int lane = tid & 31;
  const int bc = blockIdx.x;
  const int b = bc >> 6, c = bc & 63;

  const float* qrow = query + c * Esz;
  const float* crow = context + (size_t)b * Tsz * Esz;

  // Build feat = [|q-ctx|, q*ctx] (matches Wh columns 512..1023).
  for (int idx = tid; idx < Tsz * Esz; idx += 256) {
    int t = idx >> 8, e = idx & 255;
    float qv = qrow[e];
    float cv = crow[t * Esz + e];
    __bf16* fr = feat + (size_t)t * LDF;
    fr[e]       = f2bf(fabsf(qv - cv));
    fr[256 + e] = f2bf(qv * cv);
  }

  // Async-stage B panel for round 0 (rows n0..n0+31 of Wh, k 512..1023).
  __bf16* bufs[2] = {Bp0, Bp1};
  {
    for (int i = tid; i < 32 * 64; i += 256) {
      int rr = i >> 6, cc = i & 63;
      async_copy_b128((const uint4*)(Wh + (size_t)rr * K4E + KH) + cc,
                      Bp0 + (size_t)rr * LDB + cc * 8);
    }
  }
  wait_async0();
  __syncthreads();

  const int m0 = wave * 16;
  const int nl = lane & 15;
  const int rbase = (lane >> 4) << 3;
  float scorep[8];
#pragma unroll
  for (int r = 0; r < 8; ++r) scorep[r] = 0.f;

  for (int rd = 0; rd < 32; ++rd) {
    const int n0 = rd * 32;
    __bf16* cur = bufs[rd & 1];
    // Prefetch next round's B panel into the other buffer (async).
    if (rd + 1 < 32) {
      __bf16* nxt = bufs[(rd + 1) & 1];
      const int n0n = n0 + 32;
      for (int i = tid; i < 32 * 64; i += 256) {
        int rr = i >> 6, cc = i & 63;
        async_copy_b128((const uint4*)(Wh + (size_t)(n0n + rr) * K4E + KH) + cc,
                        nxt + (size_t)rr * LDB + cc * 8);
      }
    }

    v8f acc0 = {}, acc1 = {};
#pragma unroll 4
    for (int kt = 0; kt < 16; ++kt) {
      v16bf a  = load_frag_a(feat, LDF, m0, kt * 32);
      v16bf b0 = load_frag_b(cur, LDB, 0,  kt * 32);
      v16bf b1 = load_frag_b(cur, LDB, 16, kt * 32);
      acc0 = wmma_bf16(a, b0, acc0);
      acc1 = wmma_bf16(a, b1, acc1);
    }

    // Epilogue: add precomputed qW/ctxW biases, tanh, dot with Wv.
#pragma unroll
    for (int tt = 0; tt < 2; ++tt) {
      int n = n0 + tt * 16 + nl;
      float bias = bh[n] + qW[(size_t)c * K4E + n];
      float wvn = Wv[n];
      const v8f& acc = tt ? acc1 : acc0;
#pragma unroll
      for (int r = 0; r < 8; ++r) {
        int t = m0 + rbase + r;
        float pre = acc[r] + bias + ctxW[((size_t)(b * Tsz + t)) * K4E + n];
        scorep[r] += tanhf(pre) * wvn;
      }
    }
    wait_async0();     // next panel resident
    __syncthreads();   // all waves done with `cur`, `nxt` visible
  }

  // Cross-lane reduce partial scores (16 lanes per half-wave).
#pragma unroll
  for (int r = 0; r < 8; ++r) {
    float v = scorep[r];
#pragma unroll
    for (int off = 1; off < 16; off <<= 1) v += __shfl_xor(v, off, 32);
    scorep[r] = v;
  }
  if ((lane & 15) == 0) {
    int mbase = m0 + rbase;
#pragma unroll
    for (int r = 0; r < 8; ++r) sc[mbase + r] = scorep[r];
  }
  __syncthreads();

  // Masked softmax over T=128 (b_v dropped: softmax shift-invariant).
  if (tid < Tsz) {
    float s = sc[tid];
    if (mask[b * Tsz + tid] < 1) s = -1e10f;
    sc[tid] = s;
  }
  __syncthreads();
  if (tid < 32) {
    float m = fmaxf(fmaxf(sc[tid], sc[tid + 32]), fmaxf(sc[tid + 64], sc[tid + 96]));
#pragma unroll
    for (int off = 16; off; off >>= 1) m = fmaxf(m, __shfl_xor(m, off, 32));
    if (tid == 0) red[0] = m;
  }
  __syncthreads();
  float mx = red[0];
  if (tid < Tsz) sc[tid] = __expf(sc[tid] - mx);
  __syncthreads();
  if (tid < 32) {
    float s = sc[tid] + sc[tid + 32] + sc[tid + 64] + sc[tid + 96];
#pragma unroll
    for (int off = 16; off; off >>= 1) s += __shfl_xor(s, off, 32);
    if (tid == 0) red[0] = s;
  }
  __syncthreads();
  float inv = 1.0f / red[0];

  // g[b,c,e]: thread == e, coalesced ctx reads; also emit bf16 copy for gW GEMM.
  float accg = 0.f;
  for (int t = 0; t < Tsz; ++t) accg += sc[t] * crow[t * Esz + tid];
  accg *= inv;
  g_out[(size_t)bc * Esz + tid] = accg;
  gb_out[(size_t)bc * Esz + tid] = f2bf(accg);
}

// ---------------------------------------------------------------------------
// k2: h2 = tanh(qW[c]+gW[bc]+bh + [|q-g|,q*g]@WhTail^T) (LDS bf16), then
//     x = h2 @ Wlin^T + blin.  One WG per 16 (b,c)-rows; waves split N.
// ---------------------------------------------------------------------------
__global__ void __launch_bounds__(256)
k2_stage2(const float* __restrict__ query, const float* __restrict__ g_in,
          const float* __restrict__ qW, const float* __restrict__ gW,
          const __bf16* __restrict__ Wh, const float* __restrict__ bh,
          const __bf16* __restrict__ Wl, const float* __restrict__ blin,
          float* __restrict__ x_out) {
  extern __shared__ unsigned char smem[];
  __bf16* feat = (__bf16*)smem;                              // 16 x LDF
  __bf16* h2   = (__bf16*)(smem + (size_t)16 * LDF * 2);     // 16 x LDH

  const int tid = threadIdx.x, wave = tid >> 5, lane = tid & 31;
  const int m0g = blockIdx.x * 16;

  for (int idx = tid; idx < 16 * Esz; idx += 256) {
    int r = idx >> 8, e = idx & 255;
    int bc = m0g + r;
    float qv = query[(bc & 63) * Esz + e];
    float gv = g_in[(size_t)bc * Esz + e];
    __bf16* fr = feat + (size_t)r * LDF;
    fr[e]       = f2bf(fabsf(qv - gv));
    fr[256 + e] = f2bf(qv * gv);
  }
  __syncthreads();

  const int nl = lane & 15;
  const int rbase = (lane >> 4) << 3;
  // GEMM1 (K=512): 64 N-tiles across 8 waves; B from global (L2-hot).
  for (int i = 0; i < 8; ++i) {
    int n0 = (wave + 8 * i) * 16;
    v8f acc = {};
#pragma unroll 4
    for (int kt = 0; kt < 16; ++kt) {
      v16bf a  = load_frag_a(feat, LDF, 0, kt * 32);
      v16bf bb = load_frag_b(Wh + KH, K4E, n0, kt * 32);
      acc = wmma_bf16(a, bb, acc);
    }
    int n = n0 + nl;
    float bhn = bh[n];
#pragma unroll
    for (int r = 0; r < 8; ++r) {
      int bc = m0g + rbase + r;
      float pre = acc[r] + bhn + qW[(size_t)(bc & 63) * K4E + n] +
                  gW[(size_t)bc * K4E + n];
      h2[(size_t)(rbase + r) * LDH + n] = f2bf(tanhf(pre));
    }
  }
  __syncthreads();

  // GEMM2: x = h2 @ Wlin^T + blin  (K=1024, N=256 -> 2 tiles/wave)
  for (int i = 0; i < 2; ++i) {
    int n0 = (wave + 8 * i) * 16;
    v8f acc = {};
#pragma unroll 4
    for (int kt = 0; kt < 32; ++kt) {
      v16bf a  = load_frag_a(h2, LDH, 0, kt * 32);
      v16bf bb = load_frag_b(Wl, K4E, n0, kt * 32);
      acc = wmma_bf16(a, bb, acc);
    }
    int n = n0 + nl;
    float bn = blin[n];
#pragma unroll
    for (int r = 0; r < 8; ++r)
      x_out[(size_t)(m0g + rbase + r) * Esz + n] = acc[r] + bn;
  }
}

// ---------------------------------------------------------------------------
// k4: conv1d (VALID) + ReLU + max-pool. One 64-thread block per (b, conv, f).
// ---------------------------------------------------------------------------
__global__ void __launch_bounds__(64)
k4_convpool(const float* __restrict__ x,
            const float* __restrict__ w0, const float* __restrict__ c0,
            const float* __restrict__ w1, const float* __restrict__ c1,
            const float* __restrict__ w2, const float* __restrict__ c2,
            float* __restrict__ pooled) {
  int gid = blockIdx.x;
  int f  = gid & (NFsz - 1);
  int ci = (gid >> 7) % 3;
  int b  = gid / (3 * NFsz);
  int k = (ci == 0) ? 5 : (ci == 1) ? 4 : 3;
  const float* w = ((ci == 0) ? w0 : (ci == 1) ? w1 : w2) + (size_t)f * Esz * k;
  float bias = ((ci == 0) ? c0 : (ci == 1) ? c1 : c2)[f];
  int outlen = Csz - k + 1;
  int p = threadIdx.x;
  float v = 0.f;                       // relu >= 0 -> 0-pad is max-neutral
  if (p < outlen) {
    float acc = bias;
    for (int e = 0; e < Esz; ++e) {
      const float* we = w + e * k;
      const float* xe = x + ((size_t)b * Csz + p) * Esz + e;
      for (int kk = 0; kk < k; ++kk) acc += xe[(size_t)kk * Esz] * we[kk];
    }
    v = fmaxf(acc, 0.f);
  }
#pragma unroll
  for (int off = 16; off; off >>= 1) v = fmaxf(v, __shfl_xor(v, off, 32));
  __shared__ float rr[2];
  if ((threadIdx.x & 31) == 0) rr[threadIdx.x >> 5] = v;
  __syncthreads();
  if (threadIdx.x == 0) pooled[b * (3 * NFsz) + ci * NFsz + f] = fmaxf(rr[0], rr[1]);
}

// ---------------------------------------------------------------------------
// k5: out[b,o] = pooled[b,:] . W_cnn[o,:] + b_cnn[o]
// ---------------------------------------------------------------------------
__global__ void __launch_bounds__(320)
k5_final(const float* __restrict__ pooled, const float* __restrict__ Wc,
         const float* __restrict__ bcn, float* __restrict__ out) {
  int tid = threadIdx.x;
  int b = tid / 40, o = tid % 40;
  float acc = bcn[o];
  const float* p = pooled + b * 384;
  const float* w = Wc + o * 384;
  for (int j = 0; j < 384; ++j) acc += p[j] * w[j];
  out[b * 40 + o] = acc;
}

// ---------------------------------------------------------------------------
extern "C" void kernel_launch(void* const* d_in, const int* in_sizes, int n_in,
                              void* d_out, int out_size, void* d_ws, size_t ws_size,
                              hipStream_t stream) {
  (void)in_sizes; (void)n_in; (void)out_size; (void)ws_size;
  const float* query    = (const float*)d_in[0];
  const float* context  = (const float*)d_in[1];
  const int*   mask     = (const int*)d_in[2];
  const float* W_hidden = (const float*)d_in[3];
  const float* b_hidden = (const float*)d_in[4];
  const float* W_v      = (const float*)d_in[5];
  // d_in[6] = b_v: softmax shift-invariant -> no-op.
  const float* W_lin    = (const float*)d_in[7];
  const float* b_lin    = (const float*)d_in[8];
  const float* cw0 = (const float*)d_in[9];
  const float* cb0 = (const float*)d_in[10];
  const float* cw1 = (const float*)d_in[11];
  const float* cb1 = (const float*)d_in[12];
  const float* cw2 = (const float*)d_in[13];
  const float* cb2 = (const float*)d_in[14];
  const float* W_cnn = (const float*)d_in[15];
  const float* b_cnn = (const float*)d_in[16];
  float* out = (float*)d_out;

  unsigned char* ws = (unsigned char*)d_ws;
  size_t o = 0;
  __bf16* Whb  = (__bf16*)(ws + o); o += (size_t)K4E * K4E * 2;     // 2 MB
  __bf16* Wlb  = (__bf16*)(ws + o); o += (size_t)Esz * K4E * 2;     // 512 KB
  __bf16* ctxb = (__bf16*)(ws + o); o += (size_t)Bsz * Tsz * Esz * 2; // 512 KB
  __bf16* qb   = (__bf16*)(ws + o); o += (size_t)Csz * Esz * 2;     // 32 KB
  __bf16* gb   = (__bf16*)(ws + o); o += (size_t)Bsz * Csz * Esz * 2; // 256 KB
  float*  qW   = (float*)(ws + o);  o += (size_t)Csz * K4E * 4;     // 256 KB
  float*  ctxW = (float*)(ws + o);  o += (size_t)Bsz * Tsz * K4E * 4; // 4 MB
  float*  gW   = (float*)(ws + o);  o += (size_t)Bsz * Csz * K4E * 4; // 2 MB
  float*  g    = (float*)(ws + o);  o += (size_t)Bsz * Csz * Esz * 4; // 512 KB
  float*  x    = (float*)(ws + o);  o += (size_t)Bsz * Csz * Esz * 4; // 512 KB
  float*  pooled = (float*)(ws + o);                                 // 12 KB

  k0_convert<<<4096, 256, 0, stream>>>(W_hidden, W_lin, context, query,
                                       Whb, Wlb, ctxb, qb);

  // qW = q @ Wh[:, :256]^T ; ctxW = ctx @ Wh[:, 256:512]^T  (K=256)
  kgemm<<<dim3(Csz / 16, K4E / 128), 256, 0, stream>>>(qb, Esz, Whb, K4E,
                                                       qW, K4E, Esz);
  kgemm<<<dim3(Bsz * Tsz / 16, K4E / 128), 256, 0, stream>>>(ctxb, Esz,
                                                             Whb + Esz, K4E,
                                                             ctxW, K4E, Esz);

  size_t lds1 = (size_t)Tsz * LDF * 2 + (size_t)64 * LDB * 2 +
                (Tsz + 32) * sizeof(float);
  k1_attn<<<Bsz * Csz, 256, lds1, stream>>>(query, context, mask, Whb,
                                            b_hidden, W_v, qW, ctxW, g, gb);

  // gW = g @ Wh[:, 256:512]^T  (K=256)
  kgemm<<<dim3(Bsz * Csz / 16, K4E / 128), 256, 0, stream>>>(gb, Esz,
                                                             Whb + Esz, K4E,
                                                             gW, K4E, Esz);

  size_t lds2 = (size_t)16 * LDF * 2 + (size_t)16 * LDH * 2;
  k2_stage2<<<Bsz * Csz / 16, 256, lds2, stream>>>(query, g, qW, gW, Whb,
                                                   b_hidden, Wlb, b_lin, x);

  k4_convpool<<<Bsz * 3 * NFsz, 64, 0, stream>>>(x, cw0, cb0, cw1, cb1,
                                                 cw2, cb2, pooled);
  k5_final<<<1, 320, 0, stream>>>(pooled, W_cnn, b_cnn, out);
}
